// SparseMoE_16630113370886
// MI455X (gfx1250) — matrix operand order repaired and checked by reading, not compiled, not run on hardware
//
#include <hip/hip_runtime.h>
#include <hip/hip_bf16.h>

// ---------------------------------------------------------------------------
// Sparse MoE (top-2 of 8 experts, SwiGLU) for MI455X / gfx1250.
//   T=8192 tokens, D=1024, E=8, K=2, I=2752.
// router -> per-expert gather lists -> grouped WMMA GEMMs (bf16 multiplicands,
// f32 accumulate) with SwiGLU fused through LDS. Epilogues use native bf16
// converts (v_cvt_pk_bf16_f32) and v_rcp_f32 instead of IEEE division.
// ---------------------------------------------------------------------------

typedef __attribute__((ext_vector_type(16))) __bf16 v16bf;
typedef __attribute__((ext_vector_type(8)))  float  v8f;

#define MOE_T   8192
#define MOE_D   1024
#define MOE_E   8
#define MOE_I   2752
#define SX_ROW  (MOE_D + 8)     // padded LDS row stride (bf16) for x tile
#define SH_ROW  (MOE_I + 8)     // padded LDS row stride (bf16) for h tile

__device__ __forceinline__ v8f wmma_bf16(v16bf a, v16bf b, v8f c) {
    return __builtin_amdgcn_wmma_f32_16x16x32_bf16(
        /*neg_a=*/false, a, /*neg_b=*/false, b,
        /*c_mod=*/(short)0, c, /*reuse_a=*/false, /*reuse_b=*/false);
}

// A fragment (16x32 bf16, row-major in LDS). Layout: lane<16 holds row
// M=lane, K = kb..kb+15; lane>=16 same row, K = kb+16..kb+31.
// Both halves are 16B-contiguous in row-major LDS -> two ds_load_b128.
__device__ __forceinline__ v16bf load_a_frag(const __bf16* lds,
                                             int srow, int kb, int lane) {
    int m = lane & 15, h = lane >> 4;
    union { v16bf v; uint4 q[2]; } r;
    const __bf16* p = lds + m * srow + kb + 16 * h;
    r.q[0] = *(const uint4*)(p);
    r.q[1] = *(const uint4*)(p + 8);
    return r.v;
}

// B fragment (32x16, K x N) from fp32 row-major global weights (ld floats).
// Contiguous k layout used here (matches A): lane n, half h: VGPR j holds rows K = 16h+2j, 16h+2j+1, col N=n.
// Native (__bf16) casts let the compiler emit packed v_cvt_pk_bf16_f32.
__device__ __forceinline__ v16bf load_b_frag(const float* __restrict__ W,
                                             int kb, int nb, int ld, int lane) {
    int n = lane & 15, h = lane >> 4;
    union { v16bf v; __bf16 u[16]; } r;
    const float* p = W + (size_t)(kb + 16 * h) * ld + nb + n;
#pragma unroll
    for (int j = 0; j < 8; ++j) {
        r.u[2 * j]     = (__bf16)p[(size_t)(2 * j) * ld];
        r.u[2 * j + 1] = (__bf16)p[(size_t)(2 * j + 1) * ld];
    }
    return r.v;
}

// ---------------------------------------------------------------------------
// Zero output + per-expert counters (must run every launch: out is combined
// with atomics and the router atomically appends into the lists).
// ---------------------------------------------------------------------------
__global__ __launch_bounds__(256) void moe_zero(float* __restrict__ out, int n,
                                                int* __restrict__ counts) {
    int i = blockIdx.x * blockDim.x + threadIdx.x;
    if (i < MOE_E) counts[i] = 0;
    for (; i < n; i += gridDim.x * blockDim.x) out[i] = 0.0f;
}

// ---------------------------------------------------------------------------
// Router: one wave32 per token. logits = x[t] @ router_w (D x E), softmax,
// top-2, renormalize, atomic-append (token, weight) into per-expert lists.
// ---------------------------------------------------------------------------
__global__ __launch_bounds__(128) void moe_router(const float* __restrict__ x,
                                                  const float* __restrict__ rw,
                                                  int* __restrict__ counts,
                                                  int* __restrict__ lists,
                                                  float* __restrict__ lw) {
    int lane = threadIdx.x & 31;
    int t = blockIdx.x * 4 + (threadIdx.x >> 5);   // 2048 blocks * 4 waves = T

    float acc[MOE_E];
#pragma unroll
    for (int e = 0; e < MOE_E; ++e) acc[e] = 0.0f;

    const float* xp = x + (size_t)t * MOE_D;
    for (int j = 0; j < MOE_D / 32; ++j) {
        int d = j * 32 + lane;
        float xv = xp[d];
        const float* r = rw + (size_t)d * MOE_E;
#pragma unroll
        for (int e = 0; e < MOE_E; ++e) acc[e] += xv * r[e];
    }
#pragma unroll
    for (int e = 0; e < MOE_E; ++e)
        for (int off = 16; off; off >>= 1)
            acc[e] += __shfl_xor(acc[e], off, 32);

    if (lane == 0) {
        float m = acc[0];
#pragma unroll
        for (int e = 1; e < MOE_E; ++e) m = fmaxf(m, acc[e]);
        float p[MOE_E], s = 0.0f;
#pragma unroll
        for (int e = 0; e < MOE_E; ++e) { p[e] = __expf(acc[e] - m); s += p[e]; }
        float inv = __builtin_amdgcn_rcpf(s);
#pragma unroll
        for (int e = 0; e < MOE_E; ++e) p[e] *= inv;     // softmax probs

        int e0 = 0; float b0 = p[0];
#pragma unroll
        for (int e = 1; e < MOE_E; ++e) if (p[e] > b0) { b0 = p[e]; e0 = e; }
        int e1 = (e0 == 0) ? 1 : 0; float b1 = p[e1];
#pragma unroll
        for (int e = 0; e < MOE_E; ++e)
            if (e != e0 && p[e] > b1) { b1 = p[e]; e1 = e; }

        float rden = __builtin_amdgcn_rcpf(b0 + b1 + 1e-6f);
        float w0 = b0 * rden, w1 = b1 * rden;

        int p0 = atomicAdd(&counts[e0], 1);
        lists[e0 * MOE_T + p0] = t;  lw[e0 * MOE_T + p0] = w0;
        int p1 = atomicAdd(&counts[e1], 1);
        lists[e1 * MOE_T + p1] = t;  lw[e1 * MOE_T + p1] = w1;
    }
}

// ---------------------------------------------------------------------------
// Expert FFN: block = 16 gathered tokens of one expert, 256 threads (8 waves).
// Phase 1: h = silu(x @ Wg) * (x @ Wu)      (16 x I, bf16 in LDS)
// Phase 2: out[tok] += wgt * (h @ Wd)       (16 x D, atomic f32 combine)
// ---------------------------------------------------------------------------
__global__ __launch_bounds__(256) void moe_expert(const float* __restrict__ x,
                                                  const float* __restrict__ wg,
                                                  const float* __restrict__ wu,
                                                  const float* __restrict__ wd,
                                                  const int* __restrict__ counts,
                                                  const int* __restrict__ lists,
                                                  const float* __restrict__ lw,
                                                  float* __restrict__ out) {
    int e    = blockIdx.y;
    int cnt  = counts[e];
    int tile = blockIdx.x;
    if (tile * 16 >= cnt) return;               // uniform across block

    extern __shared__ __align__(16) unsigned char smem[];
    __bf16* sx   = (__bf16*)smem;                               // 16 x SX_ROW
    __bf16* sh   = sx + 16 * SX_ROW;                            // 16 x SH_ROW
    int*    stok = (int*)(sh + 16 * SH_ROW);                    // 16
    float*  swgt = (float*)(stok + 16);                         // 16

    const float* Wg = wg + (size_t)e * MOE_D * MOE_I;
    const float* Wu = wu + (size_t)e * MOE_D * MOE_I;
    const float* Wd = wd + (size_t)e * MOE_I * MOE_D;
    const int*   Le = lists + (size_t)e * MOE_T;
    const float* We = lw    + (size_t)e * MOE_T;

    // Gathered token ids + routing weights (pad rows reuse a real token, w=0).
    if (threadIdx.x < 16) {
        int idx = tile * 16 + threadIdx.x;
        int t; float w;
        if (idx < cnt) { t = Le[idx]; w = We[idx]; }
        else           { t = Le[tile * 16]; w = 0.0f; }
        stok[threadIdx.x] = t;
        swgt[threadIdx.x] = w;
    }
    __syncthreads();

    // Stage x tile (16 x 1024) as bf16 into LDS; float2 loads + paired stores.
    for (int i = threadIdx.x; i < 16 * (MOE_D / 2); i += 256) {
        int r = i >> 9, c = (i & 511) * 2;
        float2 xv = *(const float2*)(x + (size_t)stok[r] * MOE_D + c);
        __bf16* dst = sx + r * SX_ROW + c;
        dst[0] = (__bf16)xv.x;
        dst[1] = (__bf16)xv.y;
    }
    __syncthreads();

    int lane = threadIdx.x & 31;
    int wave = threadIdx.x >> 5;
    int n = lane & 15, hh = lane >> 4;

    // ---- Phase 1: gate/up projections + fused SiLU --------------------------
    for (int nt = wave; nt < MOE_I / 16; nt += 8) {
        int ib = nt * 16;
        v8f ag = {};
        v8f au = {};
#pragma unroll 2
        for (int kb = 0; kb < MOE_D; kb += 32) {
            v16bf a  = load_a_frag(sx, SX_ROW, kb, lane);
            v16bf bg = load_b_frag(Wg, kb, ib, MOE_I, lane);
            v16bf bu = load_b_frag(Wu, kb, ib, MOE_I, lane);
            if (kb + 32 < MOE_D) {   // prefetch next K-tile's weight rows
                const float* pg = Wg + (size_t)(kb + 32 + 16 * hh) * MOE_I + ib + n;
                const float* pu = Wu + (size_t)(kb + 32 + 16 * hh) * MOE_I + ib + n;
                __builtin_prefetch(pg, 0, 3);
                __builtin_prefetch(pu, 0, 3);
            }
            ag = wmma_bf16(a, bg, ag);
            au = wmma_bf16(a, bu, au);
        }
#pragma unroll
        for (int r = 0; r < 8; ++r) {               // D rows M = r + 8*hh
            int m = r + 8 * hh;
            float g = ag[r];
            float sig = __builtin_amdgcn_rcpf(1.0f + __expf(-g));  // fast sigmoid
            sh[m * SH_ROW + ib + n] = (__bf16)(g * sig * au[r]);   // silu*up
        }
    }
    __syncthreads();

    // ---- Phase 2: down projection + weighted combine ------------------------
    for (int nt = wave; nt < MOE_D / 16; nt += 8) {
        int db = nt * 16;
        v8f acc = {};
#pragma unroll 2
        for (int kb = 0; kb < MOE_I; kb += 32) {
            v16bf a = load_a_frag(sh, SH_ROW, kb, lane);
            v16bf b = load_b_frag(Wd, kb, db, MOE_D, lane);
            if (kb + 32 < MOE_I) {
                const float* pd = Wd + (size_t)(kb + 32 + 16 * hh) * MOE_D + db + n;
                __builtin_prefetch(pd, 0, 3);
            }
            acc = wmma_bf16(a, b, acc);
        }
#pragma unroll
        for (int r = 0; r < 8; ++r) {
            int m = r + 8 * hh;
            float v = swgt[m] * acc[r];
            atomicAdd(&out[(size_t)stok[m] * MOE_D + db + n], v);
        }
    }
}

// ---------------------------------------------------------------------------
extern "C" void kernel_launch(void* const* d_in, const int* in_sizes, int n_in,
                              void* d_out, int out_size, void* d_ws, size_t ws_size,
                              hipStream_t stream) {
    const float* x  = (const float*)d_in[0];   // [B,S,D] -> [T,D]
    const float* rw = (const float*)d_in[1];   // [D,E]
    const float* wg = (const float*)d_in[2];   // [E,D,I]
    const float* wu = (const float*)d_in[3];   // [E,D,I]
    const float* wd = (const float*)d_in[4];   // [E,I,D]
    float* out = (float*)d_out;                // [T,D] fp32

    unsigned char* ws = (unsigned char*)d_ws;
    int*   counts = (int*)ws;                                   // 8 ints
    int*   lists  = (int*)(ws + 256);                           // E*T ints
    float* lw     = (float*)(ws + 256 + (size_t)MOE_E * MOE_T * sizeof(int));

    (void)in_sizes; (void)n_in; (void)out_size; (void)ws_size;

    moe_zero<<<8192, 256, 0, stream>>>(out, MOE_T * MOE_D, counts);
    moe_router<<<MOE_T / 4, 128, 0, stream>>>(x, rw, counts, lists, lw);

    dim3 grid(MOE_T / 16, MOE_E);   // 512 token-tiles max per expert
    size_t lds = (size_t)16 * SX_ROW * 2 + (size_t)16 * SH_ROW * 2 + 16 * 8;
    moe_expert<<<grid, 256, lds, stream>>>(x, wg, wu, wd, counts, lists, lw, out);
}
